// MultiheadAttention_82789789598324
// MI455X (gfx1250) — matrix operand order repaired and checked by reading, not compiled, not run on hardware
//
#include <hip/hip_runtime.h>
#include <hip/hip_bf16.h>

typedef unsigned short u16;
typedef __attribute__((ext_vector_type(16))) __bf16 v16bf;
typedef __attribute__((ext_vector_type(8)))  float  v8f;
typedef __attribute__((ext_vector_type(4)))  unsigned int u32x4;
typedef __attribute__((ext_vector_type(8)))  int i32x8;
typedef __attribute__((ext_vector_type(4)))  int i32x4;

union Frag { uint4 u4[2]; v16bf v; };

static __device__ inline u16 f2bf(float f) {
    unsigned int u = __float_as_uint(f);
    unsigned int r = (u + 0x7FFFu + ((u >> 16) & 1u)) >> 16;
    return (u16)r;
}

static __device__ inline v8f wmma_bf16(v16bf a, v16bf b, v8f c) {
    // D = A(16x32 bf16) * B(32x16 bf16) + C(16x16 f32)
    return __builtin_amdgcn_wmma_f32_16x16x32_bf16(false, a, false, b, (short)0, c, false, false);
}

// flat address of a __shared__ object: addr[31:0] is the LDS byte offset
static __device__ inline unsigned lds_off(const void* p) {
    return (unsigned)(unsigned long long)p;
}

// ---- Tensor Data Mover: 2-D tile (bf16/2-byte elements) global -> LDS ----
// D# packing per CDNA5 ISA (group0 128b, group1 256b).
// This toolchain exposes the 6-arg builtin: (g0, g1, g2, g3, g4, cpol).
static __device__ inline void tdm_load_2d(unsigned lds_byte_addr, const void* gptr,
                                          unsigned tile_w,            // elements along dim0
                                          unsigned tile_h,            // rows (dim1)
                                          unsigned long long row_stride, // dim0 stride, elements
                                          unsigned td0, unsigned td1)   // tensor dims, elements
{
    unsigned long long ga = (unsigned long long)gptr;
    u32x4 g0;
    g0[0] = 1u;                                   // count=1 valid user descriptor
    g0[1] = lds_byte_addr;                        // lds_addr
    g0[2] = (unsigned)ga;                         // global_addr[31:0]
    g0[3] = (unsigned)((ga >> 32) & 0x01FFFFFFu)  // global_addr[56:32]
          | (2u << 30);                           // type = 2 ("image")
    i32x8 g1;
    g1[0] = (int)(1u << 16);                      // workgroup_mask=0, data_size=1 (2 bytes)
    g1[1] = (int)((td0 & 0xFFFFu) << 16);         // tensor_dim0[15:0]
    g1[2] = (int)(((td0 >> 16) & 0xFFFFu) | ((td1 & 0xFFFFu) << 16));
    g1[3] = (int)(((td1 >> 16) & 0xFFFFu) | ((tile_w & 0xFFFFu) << 16));
    g1[4] = (int)(tile_h & 0xFFFFu);              // tile_dim1; tile_dim2=0
    g1[5] = (int)(unsigned)(row_stride & 0xFFFFFFFFull);       // dim0_stride[31:0]
    g1[6] = (int)(unsigned)((row_stride >> 32) & 0xFFFFull);   // dim0_stride[47:32]
    g1[7] = 0;
    i32x4 z4 = {0, 0, 0, 0};
    i32x8 z8 = {0, 0, 0, 0, 0, 0, 0, 0};
    __builtin_amdgcn_tensor_load_to_lds(g0, g1, z4, z4, z8, 0);
}

// ---------------- constants ----------------
#define BATCH 2
#define SEQ   2048
#define EMB   2048
#define HEADS 16
#define HDIM  128
#define MROWS (BATCH * SEQ)   // 4096
#define NQKV  (3 * EMB)       // 6144

// ---------------- f32 -> bf16 convert ----------------
__global__ void cvt_bf16_kernel(const float* __restrict__ src, u16* __restrict__ dst, int n) {
    int i = blockIdx.x * blockDim.x + threadIdx.x;
    int stride = gridDim.x * blockDim.x;
    for (; i < n; i += stride) dst[i] = f2bf(src[i]);
}

// ---------------- tiled bf16 WMMA GEMM:  C[M,N] = A[M,K] * W[N,K]^T + bias[N] ----------------
// 256 threads (8 waves), tile 128x128, K-step 32, TDM double-buffered LDS staging.
__global__ __launch_bounds__(256) void gemm_bf16_kernel(
    const u16* __restrict__ A, const u16* __restrict__ W,
    const float* __restrict__ bias, float* __restrict__ C,
    int M, int N, int K)
{
    __shared__ __align__(16) u16 lA[2][128][32];   // [buf][m][k]
    __shared__ __align__(16) u16 lB[2][128][32];   // [buf][n][k]

    const int tid  = threadIdx.x;
    const int wave = tid >> 5, lane = tid & 31;
    const int lm = lane & 15, kg = lane >> 4;
    const int m0 = blockIdx.y * 128, n0 = blockIdx.x * 128;
    const int wm = (wave >> 1) * 32;
    const int wn = (wave & 1) * 64;

    v8f acc[2][4];
#pragma unroll
    for (int mt = 0; mt < 2; ++mt)
#pragma unroll
        for (int nt = 0; nt < 4; ++nt) acc[mt][nt] = (v8f)0.0f;

    const int nk = K / 32;
    if (wave == 0) {   // prefetch tile 0 via TDM
        tdm_load_2d(lds_off(&lA[0][0][0]), A + (size_t)m0 * K, 32, 128, K, K, M);
        tdm_load_2d(lds_off(&lB[0][0][0]), W + (size_t)n0 * K, 32, 128, K, K, N);
    }

    for (int kk = 0; kk < nk; ++kk) {
        const int p = kk & 1;
        if (wave == 0) {
            if (kk + 1 < nk) {   // prefetch next tile into other buffer, then wait current
                tdm_load_2d(lds_off(&lA[p ^ 1][0][0]),
                            A + (size_t)m0 * K + (kk + 1) * 32, 32, 128, K, K, M);
                tdm_load_2d(lds_off(&lB[p ^ 1][0][0]),
                            W + (size_t)n0 * K + (kk + 1) * 32, 32, 128, K, K, N);
                __builtin_amdgcn_s_wait_tensorcnt(2);   // 2 in flight = current tile done
            } else {
                __builtin_amdgcn_s_wait_tensorcnt(0);
            }
        }
        __syncthreads();   // tile kk resident for all waves

        Frag fb[4];
#pragma unroll
        for (int nt = 0; nt < 4; ++nt) {
            fb[nt].u4[0] = *(const uint4*)&lB[p][wn + nt * 16 + lm][8 * kg];
            fb[nt].u4[1] = *(const uint4*)&lB[p][wn + nt * 16 + lm][16 + 8 * kg];
        }
#pragma unroll
        for (int mt = 0; mt < 2; ++mt) {
            Frag fa;
            fa.u4[0] = *(const uint4*)&lA[p][wm + mt * 16 + lm][8 * kg];
            fa.u4[1] = *(const uint4*)&lA[p][wm + mt * 16 + lm][16 + 8 * kg];
#pragma unroll
            for (int nt = 0; nt < 4; ++nt)
                acc[mt][nt] = wmma_bf16(fa.v, fb[nt].v, acc[mt][nt]);
        }
        __syncthreads();   // reads done -> buffer may be overwritten by next prefetch
    }

#pragma unroll
    for (int mt = 0; mt < 2; ++mt)
#pragma unroll
        for (int nt = 0; nt < 4; ++nt)
#pragma unroll
            for (int v = 0; v < 8; ++v) {
                int r = m0 + wm + mt * 16 + v + 8 * kg;
                int cn = n0 + wn + nt * 16 + lm;
                C[(size_t)r * N + cn] = acc[mt][nt][v] + bias[cn];
            }
}

// ---------------- RoPE + full-E L2 norm + scale; split heads; emit bf16 q/k/v (B,H,S,D) ----------------
__global__ __launch_bounds__(256) void rope_norm_kernel(
    const float* __restrict__ qkvf, u16* __restrict__ qb,
    u16* __restrict__ kb, u16* __restrict__ vb)
{
    const int tok = blockIdx.x;          // b*SEQ + s
    const int b = tok >> 11, s = tok & (SEQ - 1);
    const int tid = threadIdx.x;
    __shared__ float row[EMB];
    __shared__ float red[8];
    const float lnb   = 9.210340371976184f;    // ln(10000)
    const float scale = 0.14865088937534013f;  // 2048^(-1/4)

    for (int part = 0; part < 2; ++part) {     // 0 = q, 1 = k
        const float* src = qkvf + (size_t)tok * NQKV + part * EMB;
        for (int j = tid; j < EMB; j += 256) row[j] = src[j];
        __syncthreads();

        float r[8];
        float ss = 0.f;
#pragma unroll
        for (int i = 0; i < 8; ++i) {
            int j = tid + i * 256;
            float inv = __expf(-((float)(2 * (j >> 1)) / (float)EMB) * lnb);
            float sn, cs;
            __sincosf((float)s * inv, &sn, &cs);
            float xr = (j < EMB / 2) ? -row[2 * j + 1] : row[2 * (j - EMB / 2)];
            float val = row[j] * cs + xr * sn;
            r[i] = val;
            ss += val * val;
        }
#pragma unroll
        for (int off = 16; off >= 1; off >>= 1) ss += __shfl_xor(ss, off, 32);
        if ((tid & 31) == 0) red[tid >> 5] = ss;
        __syncthreads();
        float tot = 0.f;
#pragma unroll
        for (int w = 0; w < 8; ++w) tot += red[w];
        float rn = scale / fmaxf(sqrtf(tot), 1e-12f);

        u16* dst = (part == 0) ? qb : kb;
#pragma unroll
        for (int i = 0; i < 8; ++i) {
            int j = tid + i * 256;
            int h = j >> 7, d = j & (HDIM - 1);
            dst[(((size_t)b * HEADS + h) * SEQ + s) * HDIM + d] = f2bf(r[i] * rn);
        }
        __syncthreads();
    }

    const float* vsrc = qkvf + (size_t)tok * NQKV + 2 * EMB;
    for (int j = tid; j < EMB; j += 256) {
        int h = j >> 7, d = j & (HDIM - 1);
        vb[(((size_t)b * HEADS + h) * SEQ + s) * HDIM + d] = f2bf(vsrc[j]);
    }
}

// ---------------- causal flash attention, bf16 WMMA, fp32 online softmax ----------------
// grid (SEQ/128, B*H), block 256 (8 waves); each wave owns 16 q-rows x full D=128.
// Q and K tiles are contiguous in (B,H,S,D) -> fetched by TDM; V transposed cooperatively.
__global__ __launch_bounds__(256) void attn_kernel(
    const u16* __restrict__ Qg, const u16* __restrict__ Kg,
    const u16* __restrict__ Vg, u16* __restrict__ Og)
{
    const int bh = blockIdx.y;
    const int q0 = blockIdx.x * 128;
    const size_t base = (size_t)bh * SEQ * HDIM;

    __shared__ __align__(16) u16 lQ[128][HDIM];   // 32 KB [qrow][d]
    __shared__ __align__(16) u16 lKP[64 * HDIM];  // 16 KB: K tile [kpos][d], reused as P staging
    __shared__ __align__(16) u16 lVt[HDIM][64];   // 16 KB: V^T [d][kpos]

    const int tid  = threadIdx.x;
    const int wave = tid >> 5, lane = tid & 31;
    const int lm = lane & 15, kg = lane >> 4;

    if (wave == 0)   // Q tile: 128 contiguous rows of 128 -> one 1-row TDM copy
        tdm_load_2d(lds_off(&lQ[0][0]), Qg + base + (size_t)q0 * HDIM,
                    128 * HDIM, 1, 128 * HDIM, 128 * HDIM, 1);

    v8f accO[8];
#pragma unroll
    for (int dt = 0; dt < 8; ++dt) accO[dt] = (v8f)0.0f;
    float mrow[8], lrow[8];
#pragma unroll
    for (int v = 0; v < 8; ++v) { mrow[v] = -3e38f; lrow[v] = 0.f; }

    const int nkb = (q0 >> 6) + 2;   // causal: key blocks up to diagonal
    for (int kb = 0; kb < nkb; ++kb) {
        const int k0 = kb * 64;
        __syncthreads();   // prior-iteration P/lKP reads done

        if (wave == 0)     // K tile: 64 contiguous rows of 128 -> TDM
            tdm_load_2d(lds_off(&lKP[0]), Kg + base + (size_t)k0 * HDIM,
                        64 * HDIM, 1, 64 * HDIM, 64 * HDIM, 1);

        // V^T staged cooperatively (TDM has no transpose); overlaps K DMA
#pragma unroll
        for (int i = tid; i < 64 * HDIM / 8; i += 256) {
            int r = i >> 4, c = (i & 15) * 8;
            uint4 t = *(const uint4*)&Vg[base + (size_t)(k0 + r) * HDIM + c];
            const u16* tp = (const u16*)&t;
#pragma unroll
            for (int j = 0; j < 8; ++j) lVt[c + j][r] = tp[j];
        }
        if (wave == 0) __builtin_amdgcn_s_wait_tensorcnt(0);   // Q (+K) resident
        __syncthreads();

        // S = Q K^T  (contract over d, 4 steps of 32)
        v8f sacc[4];
#pragma unroll
        for (int nt = 0; nt < 4; ++nt) sacc[nt] = (v8f)0.0f;
        const int qrb = wave * 16;
#pragma unroll
        for (int ds = 0; ds < 4; ++ds) {
            Frag fa;
            fa.u4[0] = *(const uint4*)&lQ[qrb + lm][ds * 32 + 8 * kg];
            fa.u4[1] = *(const uint4*)&lQ[qrb + lm][ds * 32 + 16 + 8 * kg];
#pragma unroll
            for (int nt = 0; nt < 4; ++nt) {
                Frag fbk;
                fbk.u4[0] = *(const uint4*)&lKP[(nt * 16 + lm) * HDIM + ds * 32 + 8 * kg];
                fbk.u4[1] = *(const uint4*)&lKP[(nt * 16 + lm) * HDIM + ds * 32 + 16 + 8 * kg];
                sacc[nt] = wmma_bf16(fa.v, fbk.v, sacc[nt]);
            }
        }
        __syncthreads();   // all waves done reading lKP -> reuse as P staging

        // online softmax update (row = v + 8*kg inside wave's 16-row strip)
        float pvals[4][8];
#pragma unroll
        for (int v = 0; v < 8; ++v) {
            const int qrow = q0 + wave * 16 + v + 8 * kg;
            float sv[4], mx = -3e38f;
#pragma unroll
            for (int nt = 0; nt < 4; ++nt) {
                int kcol = k0 + nt * 16 + lm;
                float sc = sacc[nt][v];
                if (kcol > qrow) sc = -3e38f;   // causal mask
                sv[nt] = sc;
                mx = fmaxf(mx, sc);
            }
#pragma unroll
            for (int off = 1; off < 16; off <<= 1) mx = fmaxf(mx, __shfl_xor(mx, off, 32));
            float mnew  = fmaxf(mrow[v], mx);
            float alpha = __expf(mrow[v] - mnew);
            float rs = 0.f;
#pragma unroll
            for (int nt = 0; nt < 4; ++nt) {
                float pp = __expf(sv[nt] - mnew);
                pvals[nt][v] = pp;
                rs += pp;
            }
#pragma unroll
            for (int off = 1; off < 16; off <<= 1) rs += __shfl_xor(rs, off, 32);
            mrow[v] = mnew;
            lrow[v] = lrow[v] * alpha + rs;
#pragma unroll
            for (int dt = 0; dt < 8; ++dt) accO[dt][v] *= alpha;
        }

        // stage P (bf16) into this wave's private slice of lKP, then O += P*V
        u16* P = &lKP[wave * 16 * 64];
#pragma unroll
        for (int nt = 0; nt < 4; ++nt)
#pragma unroll
            for (int v = 0; v < 8; ++v)
                P[(v + 8 * kg) * 64 + nt * 16 + lm] = f2bf(pvals[nt][v]);

#pragma unroll
        for (int ks = 0; ks < 2; ++ks) {
            Frag fa;
            fa.u4[0] = *(const uint4*)&P[lm * 64 + ks * 32 + 8 * kg];
            fa.u4[1] = *(const uint4*)&P[lm * 64 + ks * 32 + 16 + 8 * kg];
#pragma unroll
            for (int dt = 0; dt < 8; ++dt) {
                Frag fbv;
                fbv.u4[0] = *(const uint4*)&lVt[dt * 16 + lm][ks * 32 + 8 * kg];
                fbv.u4[1] = *(const uint4*)&lVt[dt * 16 + lm][ks * 32 + 16 + 8 * kg];
                accO[dt] = wmma_bf16(fa.v, fbv.v, accO[dt]);
            }
        }
    }

    // epilogue: O /= l, merge heads -> (B,S,E) bf16
    const int b = bh / HEADS, h = bh % HEADS;
#pragma unroll
    for (int dt = 0; dt < 8; ++dt)
#pragma unroll
        for (int v = 0; v < 8; ++v) {
            int row = q0 + wave * 16 + v + 8 * kg;
            int col = h * HDIM + dt * 16 + lm;
            float o = accO[dt][v] / fmaxf(lrow[v], 1e-30f);
            Og[((size_t)b * SEQ + row) * EMB + col] = f2bf(o);
        }
}

// ---------------- launcher ----------------
extern "C" void kernel_launch(void* const* d_in, const int* in_sizes, int n_in,
                              void* d_out, int out_size, void* d_ws, size_t ws_size,
                              hipStream_t stream) {
    const float* x    = (const float*)d_in[0];   // (B,S,E)
    const float* Wqkv = (const float*)d_in[1];   // (3E,E)
    const float* bqkv = (const float*)d_in[2];   // (3E)
    const float* Wout = (const float*)d_in[3];   // (E,E)
    const float* bout = (const float*)d_in[4];   // (E)
    float* out = (float*)d_out;                  // (B,S,E) fp32

    const size_t nX  = (size_t)MROWS * EMB;      // 8,388,608
    const size_t nWq = (size_t)NQKV * EMB;       // 12,582,912
    const size_t nWo = (size_t)EMB * EMB;        // 4,194,304
    u16* xb    = (u16*)d_ws;
    u16* wqkvb = xb + nX;
    u16* woutb = wqkvb + nWq;
    float* qkvf = (float*)(woutb + nWo);
    u16* qb    = (u16*)(qkvf + (size_t)MROWS * NQKV);
    u16* kb    = qb + nX;
    u16* vb    = kb + nX;
    u16* attnb = vb + nX;

    cvt_bf16_kernel<<<2048, 256, 0, stream>>>(x, xb, (int)nX);
    cvt_bf16_kernel<<<2048, 256, 0, stream>>>(Wqkv, wqkvb, (int)nWq);
    cvt_bf16_kernel<<<2048, 256, 0, stream>>>(Wout, woutb, (int)nWo);

    gemm_bf16_kernel<<<dim3(NQKV / 128, MROWS / 128), 256, 0, stream>>>(
        xb, wqkvb, bqkv, qkvf, MROWS, NQKV, EMB);

    rope_norm_kernel<<<MROWS, 256, 0, stream>>>(qkvf, qb, kb, vb);

    attn_kernel<<<dim3(SEQ / 128, BATCH * HEADS), 256, 0, stream>>>(qb, kb, vb, attnb);

    gemm_bf16_kernel<<<dim3(EMB / 128, MROWS / 128), 256, 0, stream>>>(
        attnb, woutb, bout, out, MROWS, EMB, EMB);
}
